// Conv_88321707475015
// MI455X (gfx1250) — compile-verified, matrix operand order checked by reading, not run on hardware
//
#include <hip/hip_runtime.h>
#include <hip/hip_bf16.h>
#include <stdint.h>

// ---------------------------------------------------------------------------
// Problem constants (from the reference)
// ---------------------------------------------------------------------------
#define BB     512
#define TT     30
#define NTOT   (BB * TT)          // 15360 (b,t) rows
#define HH     41
#define WF     40
#define HWF    (HH * WF)          // 1640 contiguous floats per n-row
#define NSEC   9
#define SECLEN 4
#define NF     50
#define KH     6
#define K240   (KH * WF)          // 240 = reduction length per (i,s)
#define THRESH 23.0f
#define NWIN   (NSEC * BB)        // 4608 winners

typedef _Float16     v16h __attribute__((ext_vector_type(16)));
typedef float        v8f  __attribute__((ext_vector_type(8)));
typedef unsigned int v8u  __attribute__((ext_vector_type(8)));

// ---------------------------------------------------------------------------
// A-fragment (16x32 f16, M=n-rows, K=reduction) gathered from the LDS x tile.
// Per ISA 7.12.2: lane L holds row M=L&15; VGPR0..3 = K pairs {0..7 | 8..15},
// VGPR4..7 = K {16..23 | 24..31} split by lane half. Per lane this is two
// 8-half (16B) contiguous runs -> two ds_load_b128.
// K is padded 240 -> 256; the upper run of the last K-step (k==7) is zero.
// ---------------------------------------------------------------------------
__device__ __forceinline__ v16h load_a_frag(const unsigned short* __restrict__ xs,
                                            int m, int r, int k, int hi) {
  const int base = m * HWF + r * WF + k * 32 + hi * 8;   // in halfs; 16B aligned
  uint4 lo = *(const uint4*)(xs + base);                  // K run 0..7 (+hi*8)
  uint4 hq = {0u, 0u, 0u, 0u};
  if (k < 7) hq = *(const uint4*)(xs + base + 16);        // K run 16..23 (+hi*8)
  v8u u;
  u[0] = lo.x; u[1] = lo.y; u[2] = lo.z; u[3] = lo.w;
  u[4] = hq.x; u[5] = hq.y; u[6] = hq.z; u[7] = hq.w;
  return __builtin_bit_cast(v16h, u);
}

// ---------------------------------------------------------------------------
// B-fragment (32x16 f16, K x filters). Assumed mirror of the A layout with
// lane L holding column N=L&15 and the same K packing per lane-half. Loaded
// straight from global W (L2-resident: 432 KB) with f32->f16 convert; the
// K>=240 tail (k==7 upper run) is zeroed, as are padded filters f>=50.
// ---------------------------------------------------------------------------
__device__ __forceinline__ v16h load_b_frag(const float* __restrict__ wbase,
                                            int k, bool fvalid) {
  float4 z = {0.f, 0.f, 0.f, 0.f};
  float4 a0 = z, a1 = z, b0 = z, b1 = z;
  if (fvalid) {
    const float4* p = (const float4*)(wbase + k * 32);    // 32B-aligned
    a0 = p[0]; a1 = p[1];
    if (k < 7) { b0 = p[4]; b1 = p[5]; }                  // +16 floats
  }
  v16h h;
  h[0]  = (_Float16)a0.x; h[1]  = (_Float16)a0.y;
  h[2]  = (_Float16)a0.z; h[3]  = (_Float16)a0.w;
  h[4]  = (_Float16)a1.x; h[5]  = (_Float16)a1.y;
  h[6]  = (_Float16)a1.z; h[7]  = (_Float16)a1.w;
  h[8]  = (_Float16)b0.x; h[9]  = (_Float16)b0.y;
  h[10] = (_Float16)b0.z; h[11] = (_Float16)b0.w;
  h[12] = (_Float16)b1.x; h[13] = (_Float16)b1.y;
  h[14] = (_Float16)b1.z; h[15] = (_Float16)b1.w;
  return h;
}

// ---------------------------------------------------------------------------
// Kernel 1: all 36 (section, offset) GEMMs. Block = 16 n-rows, 4 waves; wave
// w owns filter-tile w (16 filters). x tile staged once to LDS as f16.
// ---------------------------------------------------------------------------
__global__ __launch_bounds__(128) void pots_wmma_kernel(
    const float* __restrict__ x, const float* __restrict__ W,
    float* __restrict__ out) {
  __shared__ __align__(16) unsigned short xs[16 * HWF];   // 52480 B

  const int tid   = threadIdx.x;
  const int ntile = blockIdx.x;                           // 0..959

  // ---- Stage: 16 contiguous n-rows (26240 f32) -> LDS f16, packed pairs ----
  {
    const float2* xg  = (const float2*)(x + (size_t)ntile * 16 * HWF);
    unsigned int* xsu = (unsigned int*)xs;
    for (int p = tid; p < 16 * (HWF / 2); p += 128) {
      float2 v = xg[p];
      _Float16 h0 = (_Float16)v.x, h1 = (_Float16)v.y;    // x is exact in f16
      xsu[p] = (unsigned int)__builtin_bit_cast(unsigned short, h0) |
               ((unsigned int)__builtin_bit_cast(unsigned short, h1) << 16);
    }
  }
  __syncthreads();

  const int lane   = tid & 31;
  const int wave   = tid >> 5;                            // f-tile 0..3
  const int m      = lane & 15;
  const int hi     = lane >> 4;
  const int fcol   = wave * 16 + m;
  const bool fval  = fcol < NF;
  const int fclamp = fval ? fcol : (NF - 1);
  float* __restrict__ pots = out + NWIN;                  // pots after winners

  for (int i = 0; i < NSEC; ++i) {
    // B-fragments for (i, f-tile); reused across the 4 row offsets s.
    const float* wbase = W + ((size_t)(i * NF + fclamp)) * K240 + hi * 8;
    v16h Bf[8];
#pragma unroll
    for (int k = 0; k < 8; ++k) Bf[k] = load_b_frag(wbase, k, fval);

    for (int s = 0; s < SECLEN; ++s) {
      const int r = i * SECLEN + s;                       // x row 0..35
      v8f acc = {0.f, 0.f, 0.f, 0.f, 0.f, 0.f, 0.f, 0.f};
#pragma unroll
      for (int k = 0; k < 8; ++k) {
        v16h a = load_a_frag(xs, m, r, k, hi);
        acc = __builtin_amdgcn_wmma_f32_16x16x32_f16(
            /*neg_a=*/false, a, /*neg_b=*/false, Bf[k],
            /*c_mod=*/(short)0, acc, /*reuse_a=*/false, /*reuse_b=*/false);
      }
      if (fval) {
        // D layout: VGPR j -> M = j + 8*hi; pots[(i*15360+n)*200 + f*4 + s]
        float* ob = pots +
            ((size_t)i * NTOT + (size_t)ntile * 16 + hi * 8) * (NF * SECLEN) +
            (size_t)fcol * SECLEN + s;
#pragma unroll
        for (int j = 0; j < 8; ++j) ob[j * (NF * SECLEN)] = acc[j];
      }
    }
  }
}

// ---------------------------------------------------------------------------
// Kernel 2: winners. One wave per (section, batch); lane covers filters
// {lane, lane+32}. Spike masks are 30-bit words; reference logic replicated
// with popcounts + wave reductions.
// ---------------------------------------------------------------------------
__global__ __launch_bounds__(256) void winners_kernel(
    const float* __restrict__ pots, float* __restrict__ winners) {
  const int gw   = (int)((blockIdx.x * 256u + threadIdx.x) >> 5);
  const int lane = threadIdx.x & 31;
  if (gw >= NWIN) return;
  const int sec = gw / BB;
  const int b   = gw - sec * BB;
  const float* base = pots + ((size_t)sec * BB + b) * (TT * NF * SECLEN);

  unsigned int anym = 0u, mask2a[2];
  int cnta[2];
#pragma unroll
  for (int fi = 0; fi < 2; ++fi) {
    const int f = lane + fi * 32;
    unsigned int mask = 0u;
    if (f < NF) {
      for (int t = 0; t < TT; ++t) {
        float4 p = *(const float4*)(base + (t * NF + f) * SECLEN);
        float mx = fmaxf(fmaxf(p.x, p.y), fmaxf(p.z, p.w));   // max over s
        if (mx >= THRESH) mask |= (1u << t);                  // pooled[t]
      }
    }
    const int cnt = __popc(mask);
    int first = TT - cnt;                                     // clip(.,0,29)
    if (first > TT - 1) first = TT - 1;
    const unsigned int vals = (mask >> first) & 1u;
    const unsigned int m2   = vals ? mask : 0u;               // trunc[t]
    anym |= m2;
    mask2a[fi] = m2;
    cnta[fi]   = cnt;
  }
  // v = 30 * (any trunc nonzero over all t,f of this (sec,b))
  for (int d = 16; d; d >>= 1) anym |= (unsigned int)__shfl_xor((int)anym, d, 32);
  const int v = anym ? TT : 0;

  // argmax(total) with first-index (smallest f) tie-break, valid only if >0
  int key = -1;
#pragma unroll
  for (int fi = 0; fi < 2; ++fi) {
    const int f = lane + fi * 32;
    if (f < NF) {
      const int total = __popc(mask2a[fi]) + cnta[fi] * v;    // <= 930
      const int kk = (total << 8) | (255 - f);
      if (kk > key) key = kk;
    }
  }
  for (int d = 16; d; d >>= 1) {
    int o = __shfl_xor(key, d, 32);
    if (o > key) key = o;
  }
  if (lane == 0) {
    const int total = key >> 8;
    winners[gw] = (total > 0) ? (float)(255 - (key & 255)) : -1.0f;
  }
}

// ---------------------------------------------------------------------------
// Launch: inputs are x (f32, 512*30*1*41*40) and W (f32, 9*50*1*6*40).
// d_out (float) = [winners: 4608][pots: 27,648,000].
// ---------------------------------------------------------------------------
extern "C" void kernel_launch(void* const* d_in, const int* in_sizes, int n_in,
                              void* d_out, int out_size, void* d_ws, size_t ws_size,
                              hipStream_t stream) {
  const float* x = (const float*)d_in[0];
  const float* W = (const float*)d_in[1];
  float* out = (float*)d_out;

  pots_wmma_kernel<<<NTOT / 16, 128, 0, stream>>>(x, W, out);
  winners_kernel<<<(NWIN * 32) / 256, 256, 0, stream>>>(out + NWIN, out);
}
